// GraphSetAE_2250562863817
// MI455X (gfx1250) — compile-verified, matrix operand order checked by reading, not run on hardware
//
#include <hip/hip_runtime.h>

// ---------------------------------------------------------------------------
// GraphSet GAT autoencoder encoder for MI455X (gfx1250, wave32, WMMA).
// B=128, S=128, G=2000, NH=256, H=4, HD=64, NLAT=64, NGAT=2.
// Dense GEMMs use v_wmma_f32_16x16x32_bf16 (bf16 in, f32 accum) with
// register-pipelined staging + ping-pong LDS and an LDS-only split barrier
// (s_wait_dscnt + s_barrier_signal/-wait) so next-tile global loads stay in
// flight across the barrier and overlap the WMMA block.
// Attention aggregation uses v_wmma_f32_16x16x4_f32 (full f32).
// ---------------------------------------------------------------------------

#define B_    128
#define S_    128
#define G_    2000
#define NH_   256
#define H_    4
#define HD_   64
#define NLAT_ 64

typedef __attribute__((ext_vector_type(16))) __bf16        v16bf;
typedef __attribute__((ext_vector_type(8)))  float         v8f;
typedef __attribute__((ext_vector_type(2)))  float         v2f;
typedef __attribute__((ext_vector_type(4)))  unsigned int  v4u;

// LDS-only workgroup barrier: waits ds ops of this wave, then split-barrier.
// Deliberately does NOT drain LOADcnt (unlike __syncthreads), so in-flight
// global loads (register-only destinations) overlap the post-barrier WMMAs.
__device__ __forceinline__ void lds_barrier() {
    asm volatile("s_wait_dscnt 0x0\n\t"
                 "s_barrier_signal -1\n\t"
                 "s_barrier_wait -1" ::: "memory");
}

// round-to-nearest-even f32 -> bf16, two at a time, packed into one dword
__device__ __forceinline__ unsigned int pack2bf(float a, float b) {
    unsigned int ua = __float_as_uint(a);
    unsigned int ub = __float_as_uint(b);
    unsigned int ra = (ua + 0x7FFFu + ((ua >> 16) & 1u)) >> 16;
    unsigned int rb = (ub + 0x7FFFu + ((ub >> 16) & 1u)) & 0xFFFF0000u;
    return ra | rb;
}

union FragCast { v4u q[2]; v16bf v; };

// A/B fragment for 16x16x32 bf16 WMMA: lane half h reads K runs [8h,8h+8) and
// [16+8h,16+8h+8) from a 16B-aligned LDS row (row = M for A, N for B^T).
__device__ __forceinline__ v16bf load_frag_row(const unsigned short* rowp, int half) {
    FragCast f;
    f.q[0] = *reinterpret_cast<const v4u*>(rowp + 8 * half);
    f.q[1] = *reinterpret_cast<const v4u*>(rowp + 16 + 8 * half);
    return f.v;
}

// ---------------------------------------------------------------------------
// Generic C = A(MxK,f32) * B(KxN,f32) [+ bias], bf16 WMMA, f32 accumulate.
// Block: 128 threads (4 waves). Tile: BM=64, BN=64, BK=32.
// Wave w: 32x32 sub-tile (2x2 WMMA tiles) at rows 32*(w>>1), cols 32*(w&1).
// Requires: M % 64 == 0, N % 64 == 0, K % 2 == 0 (true for all call sites).
// ---------------------------------------------------------------------------
#define BK     32
#define LDSROW 40   // BK + 8 pad -> 80-byte rows, 16B aligned

struct StageRegs {
    v2f   a[8];            // A pairs (k, k+1)
    float bl[8], bh[8];    // B rows k and k+1 at column n
};

__device__ __forceinline__ void load_tile_regs(const float* __restrict__ A,
                                               const float* __restrict__ Bm,
                                               int m0, int n0, int K, int N,
                                               int k0, int tid, StageRegs& s) {
    const bool full = (k0 + BK <= K);
    #pragma unroll
    for (int r = 0; r < 8; ++r) {
        int idx = tid + (r << 7);                 // 0..1023
        int m = idx >> 4, kp = idx & 15;          // k = 2*kp
        if (full || (k0 + 2 * kp) < K) {
            s.a[r] = *reinterpret_cast<const v2f*>(
                A + (size_t)(m0 + m) * K + k0 + 2 * kp);
        } else {
            s.a[r][0] = 0.0f; s.a[r][1] = 0.0f;   // K even -> pairs never split
        }
    }
    #pragma unroll
    for (int r = 0; r < 8; ++r) {
        int idx = tid + (r << 7);
        int kp = idx >> 6, n = idx & 63;
        if (full || (k0 + 2 * kp) < K) {
            const float* bp = Bm + (size_t)(k0 + 2 * kp) * N + n0 + n;
            s.bl[r] = bp[0];
            s.bh[r] = bp[N];
        } else {
            s.bl[r] = 0.0f; s.bh[r] = 0.0f;
        }
    }
}

__device__ __forceinline__ void store_tile_lds(unsigned short (*As)[LDSROW],
                                               unsigned short (*Bs)[LDSROW],
                                               int tid, const StageRegs& s) {
    #pragma unroll
    for (int r = 0; r < 8; ++r) {
        int idx = tid + (r << 7);
        int m = idx >> 4, kp = idx & 15;
        reinterpret_cast<unsigned int*>(&As[m][0])[kp] = pack2bf(s.a[r][0], s.a[r][1]);
    }
    #pragma unroll
    for (int r = 0; r < 8; ++r) {
        int idx = tid + (r << 7);
        int kp = idx >> 6, n = idx & 63;
        reinterpret_cast<unsigned int*>(&Bs[n][0])[kp] = pack2bf(s.bl[r], s.bh[r]);
    }
}

__global__ __launch_bounds__(128)
void gemm_bf16_kernel(const float* __restrict__ A, const float* __restrict__ Bm,
                      const float* __restrict__ bias, float* __restrict__ C,
                      int M, int K, int N) {
    __shared__ unsigned short As[2][64][LDSROW];   // [buf][M][K]
    __shared__ unsigned short Bs[2][64][LDSROW];   // [buf][N][K] (B^T)
    const int tid  = threadIdx.x;
    const int wave = tid >> 5, lane = tid & 31;
    const int half = lane >> 4, lm = lane & 15;
    const int wm = wave >> 1, wn = wave & 1;
    const int m0 = blockIdx.y * 64, n0 = blockIdx.x * 64;

    v8f c00 = {}, c01 = {}, c10 = {}, c11 = {};
    const int nk = (K + BK - 1) / BK;

    StageRegs sreg;
    load_tile_regs(A, Bm, m0, n0, K, N, 0, tid, sreg);

    for (int kt = 0; kt < nk; ++kt) {
        const int p = kt & 1;
        // convert staged registers -> LDS buffer p (waits on in-flight loads)
        store_tile_lds(As[p], Bs[p], tid, sreg);
        // issue next tile's global loads; consumed at next iteration's store
        if (kt + 1 < nk)
            load_tile_regs(A, Bm, m0, n0, K, N, (kt + 1) * BK, tid, sreg);
        // LDS-only barrier: orders buf-p writes vs reads (and reads vs kt+2
        // writes) without draining LOADcnt -> loads overlap the WMMAs below.
        lds_barrier();
        v16bf a0 = load_frag_row(&As[p][32 * wm + lm][0], half);
        v16bf a1 = load_frag_row(&As[p][32 * wm + 16 + lm][0], half);
        v16bf b0 = load_frag_row(&Bs[p][32 * wn + lm][0], half);
        v16bf b1 = load_frag_row(&Bs[p][32 * wn + 16 + lm][0], half);
        c00 = __builtin_amdgcn_wmma_f32_16x16x32_bf16(false, a0, false, b0,
                                                      (short)0, c00, false, false);
        c01 = __builtin_amdgcn_wmma_f32_16x16x32_bf16(false, a0, false, b1,
                                                      (short)0, c01, false, false);
        c10 = __builtin_amdgcn_wmma_f32_16x16x32_bf16(false, a1, false, b0,
                                                      (short)0, c10, false, false);
        c11 = __builtin_amdgcn_wmma_f32_16x16x32_bf16(false, a1, false, b1,
                                                      (short)0, c11, false, false);
    }
    // epilogue: C/D layout row = r + 8*half, col = lm (coalesced stores)
    #pragma unroll
    for (int r = 0; r < 8; ++r) {
        int row0 = m0 + 32 * wm + r + 8 * half;
        int row1 = row0 + 16;
        int col0 = n0 + 32 * wn + lm;
        int col1 = col0 + 16;
        float bb0 = bias ? bias[col0] : 0.0f;
        float bb1 = bias ? bias[col1] : 0.0f;
        C[(size_t)row0 * N + col0] = c00[r] + bb0;
        C[(size_t)row0 * N + col1] = c01[r] + bb1;
        C[(size_t)row1 * N + col0] = c10[r] + bb0;
        C[(size_t)row1 * N + col1] = c11[r] + bb1;
    }
}

// ---------------------------------------------------------------------------
// Fused LayerNorm (over NH=256) + exact GELU. One block per (b,s) row.
// ---------------------------------------------------------------------------
__global__ __launch_bounds__(256)
void ln_gelu_kernel(const float* __restrict__ in, const float* __restrict__ g,
                    const float* __restrict__ b, float* __restrict__ out) {
    __shared__ float r1[256];
    __shared__ float r2[256];
    const int row = blockIdx.x, t = threadIdx.x;
    float x = in[(size_t)row * NH_ + t];
    r1[t] = x; r2[t] = x * x;
    __syncthreads();
    for (int off = 128; off >= 1; off >>= 1) {
        if (t < off) { r1[t] += r1[t + off]; r2[t] += r2[t + off]; }
        __syncthreads();
    }
    float mean = r1[0] * (1.0f / NH_);
    float var  = r2[0] * (1.0f / NH_) - mean * mean;
    float rs   = rsqrtf(var + 1e-5f);
    float y    = (x - mean) * rs * g[t] + b[t];
    out[(size_t)row * NH_ + t] = 0.5f * y * (1.0f + erff(y * 0.70710678118654752f));
}

// ---------------------------------------------------------------------------
// Per-(b,s,h): u = <hp,a_src> + <he,a_edge>,  v = <hp,a_dst> - <he,a_edge>.
// Block = 256 threads = 4 heads x 64 dims; LDS tree reduce in 64-groups.
// ---------------------------------------------------------------------------
__global__ __launch_bounds__(256)
void head_scores_kernel(const float* __restrict__ hp, const float* __restrict__ he,
                        const float* __restrict__ a_src, const float* __restrict__ a_dst,
                        const float* __restrict__ a_edge,
                        float* __restrict__ u, float* __restrict__ v) {
    __shared__ float r1[256], r2[256], r3[256];
    const int row = blockIdx.x, t = threadIdx.x;
    float p = hp[(size_t)row * NH_ + t];
    float e = he[(size_t)row * NH_ + t];
    r1[t] = p * a_src[t];
    r2[t] = p * a_dst[t];
    r3[t] = e * a_edge[t];
    __syncthreads();
    for (int off = 32; off >= 1; off >>= 1) {
        if ((t & 63) < off) {
            r1[t] += r1[t + off]; r2[t] += r2[t + off]; r3[t] += r3[t + off];
        }
        __syncthreads();
    }
    if ((t & 63) == 0) {
        int head = t >> 6;
        u[(size_t)row * H_ + head] = r1[t] + r3[t];
        v[(size_t)row * H_ + head] = r2[t] - r3[t];
    }
}

// ---------------------------------------------------------------------------
// logits[b,i,j,h] = leaky(u[b,i,h] + v[b,j,h]); mask adj==0; softmax over j.
// Output layout (B,H,S,S) so aggregation GEMM rows are contiguous.
// One block of 128 threads per (b,i,h).
// ---------------------------------------------------------------------------
__global__ __launch_bounds__(128)
void attn_softmax_kernel(const float* __restrict__ u, const float* __restrict__ v,
                         const int* __restrict__ adj, float* __restrict__ attn) {
    __shared__ float red[128];
    const int blk = blockIdx.x;
    const int b   = blk >> 9;        // / (S*H)
    const int rem = blk & 511;
    const int i   = rem >> 2, h = rem & 3;
    const int t   = threadIdx.x;     // j
    float lg = u[((size_t)b * S_ + i) * H_ + h] + v[((size_t)b * S_ + t) * H_ + h];
    lg = (lg >= 0.0f) ? lg : 0.2f * lg;               // LeakyReLU(0.2)
    if (adj[((size_t)b * S_ + i) * S_ + t] == 0) lg = -3.0e38f;
    red[t] = lg; __syncthreads();
    for (int off = 64; off >= 1; off >>= 1) {
        if (t < off) red[t] = fmaxf(red[t], red[t + off]);
        __syncthreads();
    }
    float mx = red[0];
    __syncthreads();
    float p = expf(lg - mx);                          // masked -> exp(-huge)=0
    red[t] = p; __syncthreads();
    for (int off = 64; off >= 1; off >>= 1) {
        if (t < off) red[t] += red[t + off];
        __syncthreads();
    }
    attn[((size_t)(b * H_ + h) * S_ + i) * S_ + t] = p / red[0];
}

// ---------------------------------------------------------------------------
// out[b,i,h*64+d] = sum_j attn[b,h,i,j] * hp[b,j,h*64+d]
// 512 (b,h) slices of 128x128 @ 128x64, f32 WMMA 16x16x4.
// Block: 8 waves -> 2 M-tiles x 4 N-tiles = 32x64 per block; grid (4, 512).
// Two accumulator chains break the serial C->C dependency; unroll 4 batches
// the strided hp gathers so multiple loads are in flight per wmma pair.
// ---------------------------------------------------------------------------
__global__ __launch_bounds__(256)
void aggregate_kernel(const float* __restrict__ attn, const float* __restrict__ hp,
                      float* __restrict__ out) {
    const int slice = blockIdx.y;            // b*H + h
    const int b = slice >> 2, head = slice & 3;
    const int i0 = blockIdx.x * 32;
    const int tid  = threadIdx.x;
    const int wave = tid >> 5, lane = tid & 31;
    const int half = lane >> 4, lm = lane & 15;
    const int wm = wave >> 2, wn = wave & 3;

    const float* Arow = attn + ((size_t)slice * S_ + (i0 + 16 * wm + lm)) * S_;
    const float* Bcol = hp + (size_t)b * S_ * NH_ + head * HD_ + 16 * wn + lm;

    v8f ca = {}, cb = {};
    #pragma unroll 4
    for (int k = 0; k < S_; k += 8) {
        v2f a0 = *reinterpret_cast<const v2f*>(Arow + k + 2 * half);
        v2f a1 = *reinterpret_cast<const v2f*>(Arow + k + 4 + 2 * half);
        v2f b0, b1;
        b0[0] = Bcol[(size_t)(k + 2 * half) * NH_];
        b0[1] = Bcol[(size_t)(k + 2 * half + 1) * NH_];
        b1[0] = Bcol[(size_t)(k + 4 + 2 * half) * NH_];
        b1[1] = Bcol[(size_t)(k + 4 + 2 * half + 1) * NH_];
        ca = __builtin_amdgcn_wmma_f32_16x16x4_f32(false, a0, false, b0,
                                                   (short)0, ca, false, false);
        cb = __builtin_amdgcn_wmma_f32_16x16x4_f32(false, a1, false, b1,
                                                   (short)0, cb, false, false);
    }
    v8f c = ca + cb;
    #pragma unroll
    for (int r = 0; r < 8; ++r) {
        int i = i0 + 16 * wm + r + 8 * half;
        out[((size_t)b * S_ + i) * NH_ + head * HD_ + 16 * wn + lm] = c[r];
    }
}

// ---------------------------------------------------------------------------
extern "C" void kernel_launch(void* const* d_in, const int* in_sizes, int n_in,
                              void* d_out, int out_size, void* d_ws, size_t ws_size,
                              hipStream_t stream) {
    (void)in_sizes; (void)n_in; (void)out_size; (void)ws_size;
    const float* x       = (const float*)d_in[0];
    const int*   adj     = (const int*)  d_in[1];
    const float* W_in    = (const float*)d_in[2];
    const float* b_in    = (const float*)d_in[3];
    const float* ln_in_g = (const float*)d_in[4];
    const float* ln_in_b = (const float*)d_in[5];
    const float* Wg      = (const float*)d_in[6];
    const float* a_src   = (const float*)d_in[7];
    const float* a_dst   = (const float*)d_in[8];
    const float* a_edge  = (const float*)d_in[9];
    const float* We      = (const float*)d_in[10];
    const float* ln_g    = (const float*)d_in[11];
    const float* ln_b    = (const float*)d_in[12];
    const float* Wz      = (const float*)d_in[13];
    const float* bz      = (const float*)d_in[14];
    float* out = (float*)d_out;

    const size_t HN = (size_t)B_ * S_ * NH_;       // 4,194,304 floats
    float* ws   = (float*)d_ws;
    float* h    = ws;                              // (B,S,NH)
    float* hp   = h + HN;                          // (B,S,NH)
    float* he   = hp + HN;                         // (B,S,NH) (also out buffer)
    float* u    = he + HN;                         // (B,S,H)
    float* v    = u + (size_t)B_ * S_ * H_;        // (B,S,H)
    float* attn = v + (size_t)B_ * S_ * H_;        // (B,H,S,S)

    const int MROWS = B_ * S_;                     // 16384

    // 1) input projection: hp = x @ W_in + b_in ; h = gelu(LN(hp))
    gemm_bf16_kernel<<<dim3(NH_ / 64, MROWS / 64), 128, 0, stream>>>(
        x, W_in, b_in, hp, MROWS, G_, NH_);
    ln_gelu_kernel<<<MROWS, 256, 0, stream>>>(hp, ln_in_g, ln_in_b, h);

    // 2) GAT layers
    for (int l = 0; l < 2; ++l) {
        gemm_bf16_kernel<<<dim3(NH_ / 64, MROWS / 64), 128, 0, stream>>>(
            h, Wg + (size_t)l * NH_ * NH_, nullptr, hp, MROWS, NH_, NH_);
        gemm_bf16_kernel<<<dim3(NH_ / 64, MROWS / 64), 128, 0, stream>>>(
            h, We + (size_t)l * NH_ * NH_, nullptr, he, MROWS, NH_, NH_);
        head_scores_kernel<<<MROWS, 256, 0, stream>>>(
            hp, he, a_src + l * NH_, a_dst + l * NH_, a_edge + l * NH_, u, v);
        attn_softmax_kernel<<<B_ * S_ * H_, 128, 0, stream>>>(u, v, adj, attn);
        aggregate_kernel<<<dim3(S_ / 32, B_ * H_), 256, 0, stream>>>(attn, hp, he);
        ln_gelu_kernel<<<MROWS, 256, 0, stream>>>(he, ln_g + l * NH_, ln_b + l * NH_, h);
    }

    // 3) latent projection: out = h @ Wz + bz   (N=64 -> one 64-wide block col)
    gemm_bf16_kernel<<<dim3(NLAT_ / 64, MROWS / 64), 128, 0, stream>>>(
        h, Wz, bz, out, MROWS, NH_, NLAT_);
}
